// MultiHeadedAttention_19275813225015
// MI455X (gfx1250) — compile-verified, matrix operand order checked by reading, not run on hardware
//
#include <hip/hip_runtime.h>
#include <hip/hip_bf16.h>
#include <math.h>

// ---------------------------------------------------------------------------
// MultiHeadedAttention with Gaussian focus bias — CDNA5 (gfx1250) WMMA version
// B=4, S=1024, D=1024, H=16, DK=64.
//  * one-time f32->bf16 conversion of all GEMM operands (no cvt in hot loops)
//  * all GEMM-like math via v_wmma_f32_16x16x32_bf16 (bf16 A/B, f32 acc)
//  * bf16 tile staging via global_load_async_to_lds_b128 (ASYNCcnt path)
//  * hot loop is ds_load_b128 -> v_wmma only
// ---------------------------------------------------------------------------

typedef __attribute__((ext_vector_type(16))) __bf16 v16bf;
typedef __attribute__((ext_vector_type(8)))  __bf16 v8bf;
typedef __attribute__((ext_vector_type(8)))  float  v8f;
typedef __attribute__((ext_vector_type(4)))  float  v4f;

#define WMMA_BF16(a, b, c) \
  __builtin_amdgcn_wmma_f32_16x16x32_bf16(false, (a), false, (b), (short)0, (c), false, false)

// Async global->LDS 16B copy (gfx1250; tracked by ASYNCcnt).  VDST = LDS byte
// offset (low 32 bits of a generic shared pointer, ISA 10.2), VADDR = 64-bit
// global address, SADDR = off.
__device__ __forceinline__ void async_copy_b128(unsigned int lds_byte, const void* g) {
  asm volatile("global_load_async_to_lds_b128 %0, %1, off"
               :: "v"(lds_byte), "v"(g) : "memory");
}
__device__ __forceinline__ void wait_asynccnt0() {
  asm volatile("s_wait_asynccnt 0x0" ::: "memory");
}
__device__ __forceinline__ unsigned int lds_off32(const void* p) {
  return (unsigned int)(unsigned long long)p;  // low 32 bits == LDS byte offset
}

// 16x32 bf16 A/B fragment from bf16 memory (global or LDS).  p points at this
// lane's row (row r = lane&15).  K-pattern (ISA 7.12.2): two contiguous 8-elem
// runs at K = 8*half and K = 16+8*half  ->  two 16B loads.
__device__ __forceinline__ v16bf frag_bf(const __bf16* p, int half) {
  v8bf lo = *(const v8bf*)(p + 8 * half);
  v8bf hi = *(const v8bf*)(p + 16 + 8 * half);
  v16bf f;
#pragma unroll
  for (int e = 0; e < 8; ++e) { f[e] = lo[e]; f[e + 8] = hi[e]; }
  return f;
}

// ---------------------------------------------------------------------------
// Elementwise f32 -> bf16 (8 elems/thread, b128 in / b128 out)
// ---------------------------------------------------------------------------
__global__ void cvt_bf16_kernel(const float* __restrict__ src, __bf16* __restrict__ dst,
                                size_t n) {
  size_t i = ((size_t)blockIdx.x * blockDim.x + threadIdx.x) * 8;
  if (i >= n) return;
  v4f a = *(const v4f*)(src + i);
  v4f b = *(const v4f*)(src + i + 4);
  v8bf o;
#pragma unroll
  for (int e = 0; e < 4; ++e) { o[e] = (__bf16)a[e]; o[e + 4] = (__bf16)b[e]; }
  *(v8bf*)(dst + i) = o;
}

// ---------------------------------------------------------------------------
// Tiled WMMA GEMM:  C[M,N] = A[M,K] @ W[N,K]^T + bias[N]  (+extra[b,N]) (tanh)
// A, W pre-converted bf16.  Block tile 128x128x32, 256 threads = 8 waves
// (4x2), each wave 32x64.  bf16 tiles staged to LDS by async DMA.
// Optional outputs: f32 C and/or bf16 Cbf.
// ---------------------------------------------------------------------------
__global__ __launch_bounds__(256) void gemm_wmma_kernel(
    const __bf16* __restrict__ A, const __bf16* __restrict__ W,
    const float* __restrict__ bias, const float* __restrict__ extra,
    float* __restrict__ C, __bf16* __restrict__ Cbf,
    int M, int N, int K, int bshift, int do_tanh) {
  const int TK = 32, LDT = 40;              // bf16 elems; 80B pitch (16B aligned)
  __shared__ __bf16 sA[128 * 40];
  __shared__ __bf16 sB[128 * 40];

  int tid = threadIdx.x, lane = tid & 31, wv = tid >> 5;
  int wm = wv >> 1, wn = wv & 1;            // 4x2 wave grid
  int mbase = blockIdx.x * 128, nbase = blockIdx.y * 128;
  int half = lane >> 4, r = lane & 15;

  unsigned int aOff = lds_off32(sA);
  unsigned int bOff = lds_off32(sB);

  v8f zero = {};
  v8f acc[2][4];
#pragma unroll
  for (int i = 0; i < 2; ++i)
#pragma unroll
    for (int j = 0; j < 4; ++j) acc[i][j] = zero;

  for (int k0 = 0; k0 < K; k0 += TK) {
    // Async DMA staging: 128x32 bf16 tile = 512 x 16B chunks per matrix.
    for (int p = tid; p < 512; p += 256) {
      int row = p >> 2, c8 = (p & 3) * 8;
      unsigned int lofs = (unsigned int)(row * LDT + c8) * 2u;
      async_copy_b128(aOff + lofs, &A[(size_t)(mbase + row) * K + k0 + c8]);
      async_copy_b128(bOff + lofs, &W[(size_t)(nbase + row) * K + k0 + c8]);
    }
    // One L2 prefetch per thread for the next K tile.
    if (k0 + TK < K) {
      int prow = tid >> 1;
      const __bf16* nxt = (tid & 1) ? &W[(size_t)(nbase + prow) * K + k0 + TK]
                                    : &A[(size_t)(mbase + prow) * K + k0 + TK];
      __builtin_prefetch(nxt, 0, 1);
    }
    wait_asynccnt0();
    __syncthreads();

    v16bf aF[2], bF[4];
#pragma unroll
    for (int i = 0; i < 2; ++i)
      aF[i] = frag_bf(sA + (wm * 32 + i * 16 + r) * LDT, half);
#pragma unroll
    for (int j = 0; j < 4; ++j)
      bF[j] = frag_bf(sB + (wn * 64 + j * 16 + r) * LDT, half);
#pragma unroll
    for (int i = 0; i < 2; ++i)
#pragma unroll
      for (int j = 0; j < 4; ++j) acc[i][j] = WMMA_BF16(aF[i], bF[j], acc[i][j]);
    __syncthreads();
  }

  // Epilogue. C/D layout: VGPR rr, lane L: m = rr + 8*(L>=16), n = L&15.
#pragma unroll
  for (int i = 0; i < 2; ++i)
#pragma unroll
    for (int j = 0; j < 4; ++j)
#pragma unroll
      for (int rr = 0; rr < 8; ++rr) {
        int m = mbase + wm * 32 + i * 16 + rr + 8 * half;
        int n = nbase + wn * 64 + j * 16 + r;
        float v = acc[i][j][rr] + bias[n];
        if (extra) v += extra[(m >> bshift) * N + n];
        if (do_tanh) v = tanhf(v);
        if (C)   C[(size_t)m * N + n] = v;
        if (Cbf) Cbf[(size_t)m * N + n] = (__bf16)v;
      }
}

// ---------------------------------------------------------------------------
// glo[b,d] = mean over s of q_lin[b,s,d]
// ---------------------------------------------------------------------------
__global__ void mean_seq_kernel(const float* __restrict__ q_lin, float* __restrict__ glo,
                                int B, int S, int D) {
  int idx = blockIdx.x * blockDim.x + threadIdx.x;
  if (idx >= B * D) return;
  int b = idx / D, d = idx % D;
  const float* p = q_lin + (size_t)b * S * D + d;
  float s = 0.f;
  for (int t = 0; t < S; ++t) s += p[(size_t)t * D];
  glo[idx] = s / (float)S;
}

// gbias[b,n] = glo[b,:] . W_fg[n,:] + b_fg[n]   (tiny 4xD GEMM)
__global__ void glo_proj_kernel(const float* __restrict__ glo,
                                const float* __restrict__ W_fg,
                                const float* __restrict__ b_fg,
                                float* __restrict__ gbias, int B, int D) {
  int idx = blockIdx.x * blockDim.x + threadIdx.x;
  if (idx >= B * D) return;
  int b = idx / D, n = idx % D;
  const float* g = glo + (size_t)b * D;
  const float* w = W_fg + (size_t)n * D;
  float s = b_fg[n];
  for (int k = 0; k < D; ++k) s += g[k] * w[k];
  gbias[idx] = s;
}

// P,Z[b,h,s] = sigmoid(c_head . u) * S
__global__ void pz_kernel(const float* __restrict__ c, const float* __restrict__ up,
                          const float* __restrict__ uz, float* __restrict__ P,
                          float* __restrict__ Z, int B, int S, int D, int H, int DK) {
  int idx = blockIdx.x * blockDim.x + threadIdx.x;
  if (idx >= B * H * S) return;
  int b = idx / (H * S), rem = idx % (H * S);
  int h = rem / S, s = rem % S;
  const float* cp = c + ((size_t)(b * S + s)) * D + h * DK;
  float p = 0.f, z = 0.f;
  for (int d = 0; d < DK; ++d) {
    float cv = cp[d];
    p += cv * up[h * DK + d];
    z += cv * uz[h * DK + d];
  }
  P[idx] = (1.f / (1.f + __expf(-p))) * (float)S;
  Z[idx] = (1.f / (1.f + __expf(-z))) * (float)S;
}

__global__ void zero_kernel(float* __restrict__ p, size_t n) {
  size_t i = (size_t)blockIdx.x * blockDim.x + threadIdx.x;
  if (i < n) p[i] = 0.f;
}

// ---------------------------------------------------------------------------
// Fused attention per (b, h, 16-row q tile).  4 waves / 128 threads.
// Phase 1: scores = (q @ k^T)/8 + G, masked   (WMMA from global bf16 -> 64KB LDS)
// Phase 2: softmax, probs -> bf16 LDS, accumulate attn mean (atomic, /H)
// Phase 3: x = probs @ v                      (WMMA from bf16 LDS probs)
// ---------------------------------------------------------------------------
__global__ __launch_bounds__(128) void attn_kernel(
    const __bf16* __restrict__ qbf, const __bf16* __restrict__ kbf,
    const __bf16* __restrict__ vbf, const int* __restrict__ mask,
    const float* __restrict__ P, const float* __restrict__ Z,
    __bf16* __restrict__ xbf, float* __restrict__ attn_mean,
    int B, int S, int D, int H, int DK) {
  __shared__ float  sS[16][1024];  // scores / exp scores (64 KB)
  __shared__ __bf16 sP[16][1024];  // normalized probs, bf16 (32 KB)

  int tid = threadIdx.x, lane = tid & 31, wv = tid >> 5;
  int bh = blockIdx.x;
  int b = bh / H, h = bh % H;
  int q0 = blockIdx.y * 16;
  int half = lane >> 4, r = lane & 15;

  // Per-lane P/Z for this lane's 8 output rows (m = rr + 8*half).
  float Pm[8], Zm[8];
#pragma unroll
  for (int rr = 0; rr < 8; ++rr) {
    int m = q0 + rr + 8 * half;
    Pm[rr] = P[(size_t)bh * S + m];
    Zm[rr] = Z[(size_t)bh * S + m];
  }

  const __bf16* qrow = qbf + ((size_t)(b * S + q0)) * D + h * DK + (size_t)r * D;
  v16bf aF0 = frag_bf(qrow, half);
  v16bf aF1 = frag_bf(qrow + 32, half);

  // ---- Phase 1: QK^T (+ Gaussian focus bias + mask) ----
  for (int t = wv; t < S / 16; t += 4) {   // uniform per wave
    int kp0 = t * 16;
    const __bf16* krow = kbf + ((size_t)(b * S + kp0)) * D + h * DK + (size_t)r * D;
    v16bf bF0 = frag_bf(krow, half);
    v16bf bF1 = frag_bf(krow + 32, half);
    v8f accS = {};
    accS = WMMA_BF16(aF0, bF0, accS);
    accS = WMMA_BF16(aF1, bF1, accS);
    int j = kp0 + r;                       // key position for this lane
    int mval = mask[b * S + j];
#pragma unroll
    for (int rr = 0; rr < 8; ++rr) {
      float d = (float)j - Pm[rr];
      float g = -2.f * d * d / (Zm[rr] * Zm[rr]);
      float sc = accS[rr] * 0.125f + g;    // 1/sqrt(64)
      if (mval == 0) sc = -1e15f;
      sS[rr + 8 * half][j] = sc;
    }
  }
  __syncthreads();

  // ---- Phase 2: softmax; probs -> bf16 LDS + mean-attn accumulation ----
  int row = tid >> 3, sub = tid & 7;       // 16 rows x 8 threads
  float mx = -3.4e38f;
  for (int j = sub; j < S; j += 8) mx = fmaxf(mx, sS[row][j]);
#pragma unroll
  for (int o = 1; o < 8; o <<= 1) mx = fmaxf(mx, __shfl_xor(mx, o, 32));
  float sum = 0.f;
  for (int j = sub; j < S; j += 8) {
    float e = __expf(sS[row][j] - mx);
    sS[row][j] = e;
    sum += e;
  }
#pragma unroll
  for (int o = 1; o < 8; o <<= 1) sum += __shfl_xor(sum, o, 32);
  float inv = 1.f / sum;
  float* amrow = attn_mean + ((size_t)(b * S + q0 + row)) * S;
  float hinv = 1.f / (float)H;
  for (int j = sub; j < S; j += 8) {
    float pr = sS[row][j] * inv;
    sP[row][j] = (__bf16)pr;
    atomicAdd(&amrow[j], pr * hinv);
  }
  __syncthreads();

  // ---- Phase 3: x = probs @ V (each wave one 16-feature slice) ----
  v8f accO = {};
  int n0 = wv * 16;
  for (int c0 = 0; c0 < S; c0 += 32) {
    v16bf aP = frag_bf(&sP[r][c0], half);            // probs fragment from LDS
    v16bf bV;                                        // V fragment (K-strided)
    const __bf16* vb = vbf + ((size_t)(b * S + c0)) * D + h * DK + n0 + r;
#pragma unroll
    for (int e = 0; e < 8; ++e) {
      bV[e]     = vb[(size_t)(8 * half + e) * D];
      bV[e + 8] = vb[(size_t)(16 + 8 * half + e) * D];
    }
    accO = WMMA_BF16(aP, bV, accO);
  }
#pragma unroll
  for (int rr = 0; rr < 8; ++rr)
    xbf[((size_t)(b * S + q0 + rr + 8 * half)) * D + h * DK + n0 + r] = (__bf16)accO[rr];
}

// ---------------------------------------------------------------------------
// Host-side launcher
// ---------------------------------------------------------------------------
extern "C" void kernel_launch(void* const* d_in, const int* in_sizes, int n_in,
                              void* d_out, int out_size, void* d_ws, size_t ws_size,
                              hipStream_t stream) {
  const int B = 4, S = 1024, D = 1024, H = 16, DK = 64;
  const int M = B * S, N = D, K = D;

  const float* query = (const float*)d_in[0];
  const float* key   = (const float*)d_in[1];
  const float* value = (const float*)d_in[2];
  const int*   mask  = (const int*)  d_in[3];
  const float* W_k  = (const float*)d_in[4];   const float* b_k  = (const float*)d_in[5];
  const float* W_v  = (const float*)d_in[6];   const float* b_v  = (const float*)d_in[7];
  const float* W_q  = (const float*)d_in[8];   const float* b_q  = (const float*)d_in[9];
  const float* W_o  = (const float*)d_in[10];  const float* b_o  = (const float*)d_in[11];
  const float* W_fq = (const float*)d_in[12];  const float* b_fq = (const float*)d_in[13];
  const float* W_fg = (const float*)d_in[14];  const float* b_fg = (const float*)d_in[15];
  const float* up   = (const float*)d_in[16];
  const float* uz   = (const float*)d_in[17];

  float* out       = (float*)d_out;                    // [B,S,D]
  float* attn_mean = out + (size_t)B * S * D;          // [B,S,S]

  // Workspace carve-up: f32 region first, then bf16 region.
  float* ws = (float*)d_ws;
  size_t o = 0;
  float* q_lin = ws + o; o += (size_t)M * D;           // f32 q_lin (for mean)
  float* cbuf  = ws + o; o += (size_t)M * D;           // f32 c (for P/Z)
  float* glo   = ws + o; o += (size_t)B * D;
  float* gbias = ws + o; o += (size_t)B * D;
  float* Pbuf  = ws + o; o += (size_t)B * H * S;
  float* Zbuf  = ws + o; o += (size_t)B * H * S;
  __bf16* bws = (__bf16*)(ws + o);
  size_t ob = 0;
  __bf16* qbf = bws + ob; ob += (size_t)M * D;         // bf16 q_lin
  __bf16* kbf = bws + ob; ob += (size_t)M * D;         // bf16 k
  __bf16* vbf = bws + ob; ob += (size_t)M * D;         // bf16 v
  __bf16* xbf = bws + ob; ob += (size_t)M * D;         // bf16 attention out
  __bf16* abf = bws + ob; ob += (size_t)M * D;         // bf16 staging (inputs)
  __bf16* wbf = bws + ob; ob += (size_t)N * K;         // bf16 staging (weights)
  (void)ws_size; (void)n_in; (void)in_sizes; (void)out_size;

  dim3 gGemm(M / 128, N / 128);
  const size_t nMD = (size_t)M * D, nNK = (size_t)N * K;
  const int cb = 256;
  const unsigned gMD = (unsigned)((nMD / 8 + cb - 1) / cb);
  const unsigned gNK = (unsigned)((nNK / 8 + cb - 1) / cb);

  // k = key @ W_k^T + b_k  (bf16 out only)
  cvt_bf16_kernel<<<gMD, cb, 0, stream>>>(key, abf, nMD);
  cvt_bf16_kernel<<<gNK, cb, 0, stream>>>(W_k, wbf, nNK);
  gemm_wmma_kernel<<<gGemm, 256, 0, stream>>>(abf, wbf, b_k, nullptr, nullptr, kbf,
                                              M, N, K, 10, 0);
  // v = value @ W_v^T + b_v  (bf16 out only)
  cvt_bf16_kernel<<<gMD, cb, 0, stream>>>(value, abf, nMD);
  cvt_bf16_kernel<<<gNK, cb, 0, stream>>>(W_v, wbf, nNK);
  gemm_wmma_kernel<<<gGemm, 256, 0, stream>>>(abf, wbf, b_v, nullptr, nullptr, vbf,
                                              M, N, K, 10, 0);
  // q_lin = query @ W_q^T + b_q  (f32 + bf16 out)
  cvt_bf16_kernel<<<gMD, cb, 0, stream>>>(query, abf, nMD);
  cvt_bf16_kernel<<<gNK, cb, 0, stream>>>(W_q, wbf, nNK);
  gemm_wmma_kernel<<<gGemm, 256, 0, stream>>>(abf, wbf, b_q, nullptr, q_lin, qbf,
                                              M, N, K, 10, 0);

  // Focus bias: glo = mean_s(q_lin); gbias = glo @ W_fg^T + b_fg
  mean_seq_kernel<<<(B * D + 255) / 256, 256, 0, stream>>>(q_lin, glo, B, S, D);
  glo_proj_kernel<<<(B * D + 255) / 256, 256, 0, stream>>>(glo, W_fg, b_fg, gbias, B, D);

  // c = tanh(q_lin @ W_fq^T + b_fq + gbias[b])  (f32 out only)
  cvt_bf16_kernel<<<gNK, cb, 0, stream>>>(W_fq, wbf, nNK);
  gemm_wmma_kernel<<<gGemm, 256, 0, stream>>>(qbf, wbf, b_fq, gbias, cbuf, nullptr,
                                              M, N, K, 10, 1);

  // P, Z per (b,h,s)
  pz_kernel<<<(B * H * S + 255) / 256, 256, 0, stream>>>(cbuf, up, uz, Pbuf, Zbuf,
                                                         B, S, D, H, DK);

  // Attention (+ mean over heads, accumulated atomically into d_out tail)
  size_t nmean = (size_t)B * S * S;
  zero_kernel<<<(unsigned)((nmean + 255) / 256), 256, 0, stream>>>(attn_mean, nmean);
  attn_kernel<<<dim3(B * H, S / 16), 128, 0, stream>>>(qbf, kbf, vbf, mask, Pbuf, Zbuf,
                                                       xbf, attn_mean, B, S, D, H, DK);

  // out = x @ W_o^T + b_o  (f32 out)
  cvt_bf16_kernel<<<gNK, cb, 0, stream>>>(W_o, wbf, nNK);
  gemm_wmma_kernel<<<gGemm, 256, 0, stream>>>(xbf, wbf, b_o, nullptr, out, nullptr,
                                              M, N, K, 10, 0);
}
